// HeroRecommender_73452530696630
// MI455X (gfx1250) — matrix operand order repaired and checked by reading, not compiled, not run on hardware
//
#include <hip/hip_runtime.h>

typedef __attribute__((ext_vector_type(16))) __bf16 v16bf;
typedef __attribute__((ext_vector_type(8)))  float  v8f;

#define LDS_STRIDE 40   // bf16 elements per LDS row (80B) -> breaks bank conflicts, keeps 16B align
#define BM 128
#define BN 128
#define BK 32

// ---------------------------------------------------------------------------
// Embedding gather: build X0 [8192][704] in bf16. padding_idx=0 for heroes.
// ---------------------------------------------------------------------------
__global__ __launch_bounds__(256) void gather_embed(
    const int* __restrict__ hero_ids, const int* __restrict__ attr_ids,
    const int* __restrict__ role_ids,
    const float* __restrict__ hero_emb, const float* __restrict__ attr_emb,
    const float* __restrict__ role_emb, __bf16* __restrict__ X)
{
    int row = blockIdx.x;
    for (int i = threadIdx.x; i < 704; i += blockDim.x) {
        int slot = i / 176;
        int w    = i % 176;
        float v;
        if (w < 128) {
            int id = hero_ids[row * 4 + slot];
            v = (id == 0) ? 0.0f : hero_emb[id * 128 + w];
        } else if (w < 144) {
            int id = attr_ids[row * 4 + slot];
            v = attr_emb[id * 16 + (w - 128)];
        } else {
            int id = role_ids[row * 4 + slot];
            v = role_emb[id * 32 + (w - 144)];
        }
        X[(size_t)row * 704 + i] = (__bf16)v;
    }
}

// ---------------------------------------------------------------------------
// Fold BN scale into weights, transpose to [N][K], convert to bf16.
// Wt[n*K + k] = W[k*N + n] * gamma[n] * rsqrt(var[n]+eps)
// ---------------------------------------------------------------------------
__global__ __launch_bounds__(256) void fold_weights(
    const float* __restrict__ W, const float* __restrict__ gamma,
    const float* __restrict__ rvar, __bf16* __restrict__ Wt,
    int K, int N)
{
    long long id = (long long)blockIdx.x * blockDim.x + threadIdx.x;
    long long total = (long long)K * N;
    if (id >= total) return;
    int n = (int)(id / K);
    int k = (int)(id % K);
    float s = gamma[n] * rsqrtf(rvar[n] + 1e-5f);
    Wt[id] = (__bf16)(W[(size_t)k * N + n] * s);
}

__global__ __launch_bounds__(256) void fold_bias(
    const float* __restrict__ b, const float* __restrict__ gamma,
    const float* __restrict__ beta, const float* __restrict__ rmean,
    const float* __restrict__ rvar, float* __restrict__ ob, int N)
{
    int n = blockIdx.x * blockDim.x + threadIdx.x;
    if (n >= N) return;
    float s = gamma[n] * rsqrtf(rvar[n] + 1e-5f);
    ob[n] = (b[n] - rmean[n]) * s + beta[n];
}

// ---------------------------------------------------------------------------
// Async tile loader: 128 rows x 32 cols of bf16 copied global -> LDS without
// touching VGPRs, via GLOBAL_LOAD_ASYNC_TO_LDS_B128 (ASYNCcnt-tracked).
// LDS byte address = low 32 bits of the generic pointer (ISA: addr[31:0]).
// ---------------------------------------------------------------------------
#define ASYNC_WAIT_0() asm volatile("s_wait_asynccnt 0x0" ::: "memory")

__device__ __forceinline__ void load_tile_async(const __bf16* __restrict__ g,
                                                int ld, int row0, int col0,
                                                __bf16* __restrict__ s, int tid)
{
#pragma unroll
    for (int i = 0; i < 2; ++i) {
        int idx = tid + i * 256;          // 0..511
        int r   = idx >> 2;               // 0..127
        int c   = (idx & 3) * 8;          // 0,8,16,24 (bf16 elems)
        const __bf16* gp = g + (size_t)(row0 + r) * ld + col0 + c;
        unsigned lds_addr = (unsigned)(unsigned long long)(s + r * LDS_STRIDE + c);
        asm volatile("global_load_async_to_lds_b128 %0, %1, off"
                     :: "v"(lds_addr), "v"(gp) : "memory");
    }
}

// A-fragment per ISA 16-bit A 16x32 layout:
//   lanes 0-15: elems 0..7 -> K=0..7,  elems 8..15 -> K=16..23  (M = lane)
//   lanes16-31: elems 0..7 -> K=8..15, elems 8..15 -> K=24..31  (M = lane-16)
__device__ __forceinline__ v16bf load_a_frag(const __bf16* __restrict__ s,
                                             int mbase, int lane)
{
    int m    = mbase + (lane & 15);
    int half = lane >> 4;
    const __bf16* p = s + m * LDS_STRIDE + half * 8;   // k = half*8
    union { uint4 u[2]; v16bf v; } t;
    t.u[0] = *reinterpret_cast<const uint4*>(p);        // K = half*8 + 0..7
    t.u[1] = *reinterpret_cast<const uint4*>(p + 16);   // K = half*8 + 16..23
    return t.v;
}

// B-fragment per ISA 16-bit B 32x16 layout (LDS holds W transposed, n-major):
//   lanes 0-15: K=0..7,16..23 for N=lane; lanes 16-31: K=8..15,24..31 for N=lane-16
__device__ __forceinline__ v16bf load_b_frag(const __bf16* __restrict__ s,
                                             int nbase, int lane)
{
    int n    = nbase + (lane & 15);
    int half = lane >> 4;
    const __bf16* p = s + n * LDS_STRIDE + half * 8;   // k = half*8
    union { uint4 u[2]; v16bf v; } t;
    t.u[0] = *reinterpret_cast<const uint4*>(p);        // K = half*8 + 0..7
    t.u[1] = *reinterpret_cast<const uint4*>(p + 16);   // K = half*8 + 16..23
    return t.v;
}

// ---------------------------------------------------------------------------
// Fused GEMM + folded-BN bias + LeakyReLU.  out[M][N] (bf16) = lrelu(A @ Wt^T + bias)
// A: [M][K] bf16 row-major, Wt: [N][K] bf16 row-major. 128x128 block tile,
// 8 wave32s, each wave 32x64 (2x4 wmma tiles), BK=32, double-buffered LDS
// filled by async global->LDS DMA.
// ---------------------------------------------------------------------------
__global__ __launch_bounds__(256) void gemm_bn_lrelu(
    const __bf16* __restrict__ A, const __bf16* __restrict__ Wt,
    const float* __restrict__ bias, __bf16* __restrict__ out,
    int M, int N, int K)
{
    __shared__ __bf16 sA[2][BM * LDS_STRIDE];
    __shared__ __bf16 sB[2][BN * LDS_STRIDE];

    const int tid  = threadIdx.x;
    const int lane = tid & 31;
    const int wave = tid >> 5;
    const int wm   = wave >> 1;   // 0..3  -> 32-row strip
    const int wn   = wave & 1;    // 0..1  -> 64-col strip
    const int m0   = blockIdx.y * BM;
    const int n0   = blockIdx.x * BN;

    v8f acc[2][4];
#pragma unroll
    for (int mi = 0; mi < 2; ++mi)
#pragma unroll
        for (int ni = 0; ni < 4; ++ni)
#pragma unroll
            for (int e = 0; e < 8; ++e) acc[mi][ni][e] = 0.0f;

    const int nk = K / BK;
    load_tile_async(A,  K, m0, 0, sA[0], tid);
    load_tile_async(Wt, K, n0, 0, sB[0], tid);
    ASYNC_WAIT_0();
    __syncthreads();

    for (int kt = 0; kt < nk; ++kt) {
        const int cur = kt & 1;
        const int nxt = cur ^ 1;
        if (kt + 1 < nk) {   // kick off next tile's async DMA into the other buffer
            load_tile_async(A,  K, m0, (kt + 1) * BK, sA[nxt], tid);
            load_tile_async(Wt, K, n0, (kt + 1) * BK, sB[nxt], tid);
        }
        if (kt + 2 < nk) {   // pull K+2 tiles toward L2 (global_prefetch_b8)
            __builtin_prefetch(A  + (size_t)(m0 + (tid >> 1)) * K + (kt + 2) * BK, 0, 1);
            __builtin_prefetch(Wt + (size_t)(n0 + (tid >> 1)) * K + (kt + 2) * BK, 0, 1);
        }

        v16bf af[2], bf[4];
#pragma unroll
        for (int mi = 0; mi < 2; ++mi)
            af[mi] = load_a_frag(sA[cur], wm * 32 + mi * 16, lane);
#pragma unroll
        for (int ni = 0; ni < 4; ++ni)
            bf[ni] = load_b_frag(sB[cur], wn * 64 + ni * 16, lane);

#pragma unroll
        for (int mi = 0; mi < 2; ++mi)
#pragma unroll
            for (int ni = 0; ni < 4; ++ni)
                acc[mi][ni] = __builtin_amdgcn_wmma_f32_16x16x32_bf16(
                    false, af[mi], false, bf[ni], (short)0, acc[mi][ni],
                    false, false);

        if (kt + 1 < nk) ASYNC_WAIT_0();   // our async writes to buf[nxt] landed
        __syncthreads();                   // everyone's writes visible, reads done
    }

    // Epilogue: C/D layout -> VGPR r holds M = r + 8*(lane>>4), N = lane&15.
    const int half = lane >> 4;
    const int nl   = lane & 15;
#pragma unroll
    for (int mi = 0; mi < 2; ++mi) {
#pragma unroll
        for (int ni = 0; ni < 4; ++ni) {
            int ncol = n0 + wn * 64 + ni * 16 + nl;
            float bs = bias[ncol];
#pragma unroll
            for (int r = 0; r < 8; ++r) {
                int mrow = m0 + wm * 32 + mi * 16 + half * 8 + r;
                float v = acc[mi][ni][r] + bs;
                v = (v > 0.0f) ? v : 0.01f * v;
                out[(size_t)mrow * N + ncol] = (__bf16)v;
            }
        }
    }
}

// ---------------------------------------------------------------------------
// Decoder: scores[8192][130] = X @ dec_w + dec_b, then hero mask / col 0 fill.
// ---------------------------------------------------------------------------
__global__ __launch_bounds__(160) void decoder_mask(
    const __bf16* __restrict__ X, const float* __restrict__ Wd,
    const float* __restrict__ bd, const int* __restrict__ hero_ids,
    float* __restrict__ out)
{
    __shared__ float xs[1408];
    int row = blockIdx.x;
    for (int i = threadIdx.x; i < 1408; i += blockDim.x)
        xs[i] = (float)X[(size_t)row * 1408 + i];
    __syncthreads();

    int h = threadIdx.x;
    if (h < 130) {
        float acc = bd[h];
        for (int k = 0; k < 1408; ++k)
            acc = fmaf(xs[k], Wd[k * 130 + h], acc);
        bool mask = (h == 0);
#pragma unroll
        for (int j = 0; j < 4; ++j) {
            int id = hero_ids[row * 4 + j];
            mask = mask || (id > 0 && id == h);
        }
        out[(size_t)row * 130 + h] = mask ? -100.0f : acc;
    }
}

// ---------------------------------------------------------------------------
static inline size_t align256(size_t x) { return (x + 255) & ~(size_t)255; }

extern "C" void kernel_launch(void* const* d_in, const int* in_sizes, int n_in,
                              void* d_out, int out_size, void* d_ws, size_t ws_size,
                              hipStream_t stream)
{
    (void)in_sizes; (void)n_in; (void)out_size; (void)ws_size;
    static const int DIMS[8] = {704, 1408, 2816, 5632, 11264, 5632, 2816, 1408};
    const int BATCH = 8192;

    const int*   hero_ids = (const int*)d_in[0];
    const int*   attr_ids = (const int*)d_in[1];
    const int*   role_ids = (const int*)d_in[2];
    const float* hero_emb = (const float*)d_in[3];
    const float* attr_emb = (const float*)d_in[4];
    const float* role_emb = (const float*)d_in[5];
    // enc_params: 7 layers x (W, b, gamma, beta, rmean, rvar) starting at index 6
    const float* dec_w = (const float*)d_in[48];
    const float* dec_b = (const float*)d_in[49];
    float* out = (float*)d_out;

    // Workspace layout
    char* ws = (char*)d_ws;
    size_t off = 0;
    size_t wt_off[7], bias_off[7];
    for (int i = 0; i < 7; ++i) {
        wt_off[i] = off;
        off = align256(off + (size_t)DIMS[i] * DIMS[i + 1] * sizeof(__bf16));
    }
    for (int i = 0; i < 7; ++i) {
        bias_off[i] = off;
        off = align256(off + (size_t)DIMS[i + 1] * sizeof(float));
    }
    size_t act_bytes = (size_t)BATCH * 11264 * sizeof(__bf16);
    size_t actA_off = off; off = align256(off + act_bytes);
    size_t actB_off = off; off = align256(off + act_bytes);

    __bf16* actA = (__bf16*)(ws + actA_off);
    __bf16* actB = (__bf16*)(ws + actB_off);

    // 1) Build X0 (bf16) in actA
    gather_embed<<<BATCH, 256, 0, stream>>>(hero_ids, attr_ids, role_ids,
                                            hero_emb, attr_emb, role_emb, actA);

    // 2) Fold BN into bf16 transposed weights + fp32 biases
    for (int i = 0; i < 7; ++i) {
        const float* W     = (const float*)d_in[6 + i * 6 + 0];
        const float* b     = (const float*)d_in[6 + i * 6 + 1];
        const float* gamma = (const float*)d_in[6 + i * 6 + 2];
        const float* beta  = (const float*)d_in[6 + i * 6 + 3];
        const float* rmean = (const float*)d_in[6 + i * 6 + 4];
        const float* rvar  = (const float*)d_in[6 + i * 6 + 5];
        int K = DIMS[i], N = DIMS[i + 1];
        long long total = (long long)K * N;
        int blocks = (int)((total + 255) / 256);
        fold_weights<<<blocks, 256, 0, stream>>>(W, gamma, rvar,
                                                 (__bf16*)(ws + wt_off[i]), K, N);
        fold_bias<<<(N + 255) / 256, 256, 0, stream>>>(b, gamma, beta, rmean, rvar,
                                                       (float*)(ws + bias_off[i]), N);
    }

    // 3) 7 fused GEMM+BN+LeakyReLU layers, ping-ponging actA/actB
    __bf16* cur = actA;
    __bf16* nxt = actB;
    for (int i = 0; i < 7; ++i) {
        int K = DIMS[i], N = DIMS[i + 1];
        dim3 grid(N / BN, BATCH / BM);
        gemm_bn_lrelu<<<grid, 256, 0, stream>>>(cur, (const __bf16*)(ws + wt_off[i]),
                                                (const float*)(ws + bias_off[i]),
                                                nxt, BATCH, N, K);
        __bf16* t = cur; cur = nxt; nxt = t;
    }
    // after 7 swaps, `cur` holds the final 1408-wide activations

    // 4) Decoder + mask
    decoder_mask<<<BATCH, 160, 0, stream>>>(cur, dec_w, dec_b, hero_ids, out);
}
